// MultiHeadLatentAttention_20770461843735
// MI455X (gfx1250) — compile-verified, matrix operand order checked by reading, not run on hardware
//
#include <hip/hip_runtime.h>
#include <hip/hip_bf16.h>
#include <cstdint>
#include <cstddef>

typedef __bf16 bf16;
typedef bf16 v16bf __attribute__((ext_vector_type(16)));
typedef bf16 bf16x2 __attribute__((ext_vector_type(2)));
typedef float v8f  __attribute__((ext_vector_type(8)));
typedef unsigned int u32x4 __attribute__((ext_vector_type(4)));
typedef int i32x4 __attribute__((ext_vector_type(4)));
typedef int i32x8 __attribute__((ext_vector_type(8)));

// CDNA5 16-bit A-matrix (16x32) element -> K mapping.
// Lane L: row = L%16, khalf = L/16. Vector element e (pairs contiguous in K):
//   e in [0,8):  K = khalf*8 + e
//   e in [8,16): K = 16 + khalf*8 + (e-8)
__device__ __forceinline__ int kmap(int e, int khalf) {
  return (e < 8) ? (khalf * 8 + e) : (16 + khalf * 8 + (e - 8));
}

__device__ __forceinline__ v8f wmma_bf16(v16bf a, v16bf b, v8f c) {
  return __builtin_amdgcn_wmma_f32_16x16x32_bf16(false, a, false, b, (short)0, c,
                                                 false, false);
}

__device__ __forceinline__ unsigned lds_off(const void* p) {
  // Flat shared pointer: bits [31:0] are the LDS byte offset (ISA 10.2).
  return (unsigned)(uintptr_t)p;
}

// ---------------------------------------------------------------------------
// Issue a TDM 2D tile load: tile (tile_d1 rows x tile_d0 elems, 2-byte
// elements, rows separated by stride0 elems in memory) -> LDS at lds_byte,
// packed row-major (row y at lds_byte + y*tile_d0*2).
// D# layout per CDNA5 ISA (groups 0 and 1; 2D => groups
// 2/3 zero).
// ---------------------------------------------------------------------------
__device__ __forceinline__ void tdm_load_2d(unsigned lds_byte, const bf16* gptr,
                                            unsigned tile_d0, unsigned tile_d1,
                                            unsigned long tensor_d0,
                                            unsigned long tensor_d1,
                                            unsigned long stride0) {
  unsigned long ga = (unsigned long)(uintptr_t)gptr;
  u32x4 g0;
  g0[0] = 1u;                                       // count=1 (valid), user mode
  g0[1] = lds_byte;                                 // lds_addr (bytes)
  g0[2] = (unsigned)(ga & 0xffffffffu);             // global_addr[31:0]
  g0[3] = (unsigned)((ga >> 32) & 0x01ffffffu)      // global_addr[56:32]
          | (2u << 30);                             // type = 2 ("image")
  i32x8 g1;
  g1[0] = (int)(1u << 16);                          // data_size=1 => 2 bytes
  g1[1] = (int)((tensor_d0 & 0xffffu) << 16);       // tensor_dim0[15:0]
  g1[2] = (int)(((tensor_d0 >> 16) & 0xffffu) |     // tensor_dim0[31:16]
                ((tensor_d1 & 0xffffu) << 16));     // tensor_dim1[15:0]
  g1[3] = (int)(((tensor_d1 >> 16) & 0xffffu) |     // tensor_dim1[31:16]
                ((tile_d0 & 0xffffu) << 16));       // tile_dim0
  g1[4] = (int)(tile_d1 & 0xffffu);                 // tile_dim1 (tile_dim2=0)
  g1[5] = (int)(unsigned)(stride0 & 0xffffffffu);   // tensor_dim0_stride[31:0]
  g1[6] = (int)(unsigned)((stride0 >> 32) & 0xffffu); // stride[47:32], stride1=0
  g1[7] = 0;
  i32x4 z4 = {0, 0, 0, 0};
#if defined(__clang_major__) && __clang_major__ >= 23
  i32x8 z8 = {0, 0, 0, 0, 0, 0, 0, 0};
  __builtin_amdgcn_tensor_load_to_lds(g0, g1, z4, z4, z8, 0);
#else
  __builtin_amdgcn_tensor_load_to_lds(g0, g1, z4, z4, 0);
#endif
}

// ---------------------------------------------------------------------------
// Operand packing (run once; weights are reused 256x so pre-packing to bf16,
// and transposing weights to N x K, makes every GEMM fragment a contiguous
// 16B load).
// ---------------------------------------------------------------------------
__global__ void pack_bf16_kernel(const float* __restrict__ in,
                                 bf16* __restrict__ out, int n) {
  int i = blockIdx.x * blockDim.x + threadIdx.x;
  if (i < n) out[i] = (bf16)in[i];
}

__global__ void pack_wt_kernel(const float* __restrict__ W,
                               bf16* __restrict__ Wt, int K, int N) {
  int i = blockIdx.x * blockDim.x + threadIdx.x;  // i = k*N + n (coalesced read)
  if (i >= K * N) return;
  int k = i / N, n = i % N;
  Wt[(size_t)n * K + k] = (bf16)W[i];
}

// ---------------------------------------------------------------------------
// GEMM: C[M,N] = A[M,K] @ Bt[N,K]^T, bf16 operands, f32 accumulate.
// Block = 128 threads (4 waves) computing a 32(M) x 64(N) tile.
// Wave 0 drives the Tensor Data Mover: per 64-deep K step it issues two 2D
// TDM descriptors (A 32x64, B 64x64) into double-buffered LDS, overlapped via
// TENSORcnt (wait <= 2 leaves the next buffer's pair in flight).
// Each wave computes 16x32: A fragment reused across 2 WMMAs, 4 WMMAs/step.
// ---------------------------------------------------------------------------
template <bool OUT_BF16, bool SPLIT, bool BIAS>
__global__ __launch_bounds__(128) void gemm_wmma_tdm_kernel(
    const bf16* __restrict__ A, const bf16* __restrict__ Bt,
    const float* __restrict__ bias, void* __restrict__ Cp, int M, int N, int K,
    int ldc, int groupW, int colOff) {
  __shared__ __align__(16) bf16 Alds[2][32 * 64];  // 2 x 4 KB
  __shared__ __align__(16) bf16 Blds[2][64 * 64];  // 2 x 8 KB

  const int tid = threadIdx.x;
  const int lane = tid & 31;
  const int wave = tid >> 5;  // 0..3
  const int wm = wave >> 1;   // m-subtile 0..1
  const int wn = wave & 1;    // n-subtile 0..1
  const int hl = lane >> 4;
  const int cn = lane & 15;
  const int m0 = blockIdx.y * 32;
  const int n0 = blockIdx.x * 64;

  v8f acc[2];
#pragma unroll
  for (int t = 0; t < 2; ++t)
#pragma unroll
    for (int r = 0; r < 8; ++r) acc[t][r] = 0.0f;

  const int nIter = K / 64;

  if (wave == 0) {  // prologue: fill buffer 0
    tdm_load_2d(lds_off(&Alds[0][0]), A + (size_t)m0 * K, 64, 32, K, M, K);
    tdm_load_2d(lds_off(&Blds[0][0]), Bt + (size_t)n0 * K, 64, 64, K, N, K);
  }

  for (int i = 0; i < nIter; ++i) {
    const int cur = i & 1;
    const int nxt = cur ^ 1;
    if (wave == 0) {
      if (i + 1 < nIter) {
        const int kb = (i + 1) * 64;
        tdm_load_2d(lds_off(&Alds[nxt][0]), A + (size_t)m0 * K + kb, 64, 32, K,
                    M, K);
        tdm_load_2d(lds_off(&Blds[nxt][0]), Bt + (size_t)n0 * K + kb, 64, 64, K,
                    N, K);
        __builtin_amdgcn_s_wait_tensorcnt(2);  // current buffer's pair done
      } else {
        __builtin_amdgcn_s_wait_tensorcnt(0);
      }
    }
    __syncthreads();

    const bf16* At = &Alds[cur][0];
    const bf16* Bl = &Blds[cur][0];
#pragma unroll
    for (int kc = 0; kc < 2; ++kc) {
      v16bf a;
#pragma unroll
      for (int e = 0; e < 16; e += 2) {
        int kk = kc * 32 + kmap(e, hl);
        bf16x2 p = *reinterpret_cast<const bf16x2*>(At + (wm * 16 + cn) * 64 + kk);
        a[e] = p.x;
        a[e + 1] = p.y;
      }
#pragma unroll
      for (int t = 0; t < 2; ++t) {
        v16bf bb;
#pragma unroll
        for (int e = 0; e < 16; e += 2) {
          int kk = kc * 32 + kmap(e, hl);
          bf16x2 p = *reinterpret_cast<const bf16x2*>(
              Bl + (wn * 32 + t * 16 + cn) * 64 + kk);
          bb[e] = p.x;
          bb[e + 1] = p.y;
        }
        acc[t] = wmma_bf16(a, bb, acc[t]);
      }
    }
    __syncthreads();  // all waves done reading before buffer is refilled
  }

#pragma unroll
  for (int t = 0; t < 2; ++t) {
#pragma unroll
    for (int r = 0; r < 8; ++r) {
      int m = m0 + wm * 16 + r + 8 * hl;
      int n = n0 + wn * 32 + t * 16 + cn;
      float val = acc[t][r];
      if (BIAS) val += bias[n];
      size_t oidx;
      if (SPLIT)
        oidx = (size_t)m * ldc + (size_t)(n / groupW) * 128 + (n % groupW) +
               colOff;
      else
        oidx = (size_t)m * ldc + n;
      if (OUT_BF16)
        ((bf16*)Cp)[oidx] = (bf16)val;
      else
        ((float*)Cp)[oidx] = val;
    }
  }
}

// ---------------------------------------------------------------------------
// RoPE for q: q layout (B,S,H,128); rotates d in [64,128).
// ---------------------------------------------------------------------------
__global__ void rope_q_kernel(bf16* __restrict__ q, int total, int S) {
  int i = blockIdx.x * blockDim.x + threadIdx.x;
  if (i >= total) return;
  int p = i & 31;
  int row = i >> 5;  // (b*S + s)*16 + h
  int s = (row >> 4) % S;
  float inv = __powf(10000.0f, -(float)(2 * p) * (1.0f / 64.0f));
  float ang = (float)s * inv;
  float c, sn;
  __sincosf(ang, &sn, &c);
  bf16* ptr = q + (size_t)row * 128 + 64;
  float x1 = (float)ptr[p];
  float x2 = (float)ptr[p + 32];
  ptr[p] = (bf16)(x1 * c - x2 * sn);
  ptr[p + 32] = (bf16)(x2 * c + x1 * sn);
}

// RoPE for k_r (B,S,64) and broadcast into k[...,64:128] for all 16 heads.
__global__ void rope_k_kernel(const bf16* __restrict__ kr, bf16* __restrict__ k,
                              int total, int S) {
  int i = blockIdx.x * blockDim.x + threadIdx.x;
  if (i >= total) return;
  int p = i & 31;
  int bs = i >> 5;  // b*S + s
  int s = bs % S;
  float inv = __powf(10000.0f, -(float)(2 * p) * (1.0f / 64.0f));
  float ang = (float)s * inv;
  float c, sn;
  __sincosf(ang, &sn, &c);
  float x1 = (float)kr[(size_t)bs * 64 + p];
  float x2 = (float)kr[(size_t)bs * 64 + p + 32];
  bf16 r1 = (bf16)(x1 * c - x2 * sn);
  bf16 r2 = (bf16)(x2 * c + x1 * sn);
  bf16* base = k + (size_t)bs * 2048 + 64;
#pragma unroll
  for (int h = 0; h < 16; ++h) {
    base[h * 128 + p] = r1;
    base[h * 128 + 32 + p] = r2;
  }
}

// ---------------------------------------------------------------------------
// Flash attention: one wave per (b, h, 16-row q tile). KV streamed in chunks
// of 32. QK^T and P*V via bf16 WMMA, online softmax with wave32 shuffles.
// q/k/v/ctx layout: (B, S, H, 128) bf16.
// ---------------------------------------------------------------------------
__global__ __launch_bounds__(32) void mla_attn_kernel(
    const bf16* __restrict__ q, const bf16* __restrict__ k,
    const bf16* __restrict__ v, bf16* __restrict__ ctx, int S) {
  __shared__ __align__(16) bf16 Vlds[32 * 128];  // V chunk (kpos, d)
  __shared__ float Plds[16 * 32];                // P tile for C->A transpose

  const int lane = threadIdx.x;
  const int hl = lane >> 4;
  const int cn = lane & 15;
  const int q0 = blockIdx.x * 16;
  const int h = blockIdx.y;
  const int b = blockIdx.z;
  const float scale = 0.0883883476483184f;  // 1/sqrt(128)

  v16bf Qa[4];
  {
    const bf16* qrow = q + (size_t)((b * S + q0 + cn) * 16 + h) * 128;
#pragma unroll
    for (int dc = 0; dc < 4; ++dc) {
#pragma unroll
      for (int e = 0; e < 16; e += 2) {
        int d = dc * 32 + kmap(e, hl);
        bf16x2 p = *reinterpret_cast<const bf16x2*>(qrow + d);
        Qa[dc][e] = p.x;
        Qa[dc][e + 1] = p.y;
      }
    }
  }

  v8f acc[8];
#pragma unroll
  for (int dt = 0; dt < 8; ++dt)
#pragma unroll
    for (int r = 0; r < 8; ++r) acc[dt][r] = 0.0f;
  float m_i[8], l_i[8];
#pragma unroll
  for (int r = 0; r < 8; ++r) {
    m_i[r] = -1e30f;
    l_i[r] = 0.0f;
  }

  const int jmax = (q0 + 15) / 32;
  for (int j = 0; j <= jmax; ++j) {
    const int kv0 = j * 32;
    __syncthreads();  // protect Vlds/Plds from previous iteration's readers

    {  // stage V chunk (32 x 128 bf16), one kpos row per lane
      const bf16* vrow = v + (size_t)((b * S + kv0 + lane) * 16 + h) * 128;
      const int4* src = reinterpret_cast<const int4*>(vrow);
      int4* dst = reinterpret_cast<int4*>(&Vlds[lane * 128]);
#pragma unroll
      for (int t = 0; t < 16; ++t) dst[t] = src[t];
    }

    // scores: two 16x16 tiles, each sum of 4 WMMAs over d=128
    v8f sc[2];
#pragma unroll
    for (int t = 0; t < 2; ++t) {
#pragma unroll
      for (int r = 0; r < 8; ++r) sc[t][r] = 0.0f;
      const bf16* krow =
          k + (size_t)((b * S + kv0 + t * 16 + cn) * 16 + h) * 128;
#pragma unroll
      for (int dc = 0; dc < 4; ++dc) {
        v16bf Kb;
#pragma unroll
        for (int e = 0; e < 16; e += 2) {
          int d = dc * 32 + kmap(e, hl);
          bf16x2 p = *reinterpret_cast<const bf16x2*>(krow + d);
          Kb[e] = p.x;
          Kb[e + 1] = p.y;
        }
        sc[t] = wmma_bf16(Qa[dc], Kb, sc[t]);
      }
    }

    // causal mask + scale; per-row chunk max
    float mx[8], alpha[8], rs[8];
#pragma unroll
    for (int r = 0; r < 8; ++r) {
      int rowg = q0 + r + 8 * hl;
      float s0 = sc[0][r] * scale;
      float s1 = sc[1][r] * scale;
      if (kv0 + cn > rowg) s0 = -1e30f;
      if (kv0 + 16 + cn > rowg) s1 = -1e30f;
      sc[0][r] = s0;
      sc[1][r] = s1;
      mx[r] = fmaxf(s0, s1);
    }
#pragma unroll
    for (int off = 1; off < 16; off <<= 1)
#pragma unroll
      for (int r = 0; r < 8; ++r)
        mx[r] = fmaxf(mx[r], __shfl_xor(mx[r], off, 32));

#pragma unroll
    for (int r = 0; r < 8; ++r) {
      float mnew = fmaxf(m_i[r], mx[r]);
      alpha[r] = __expf(m_i[r] - mnew);
      m_i[r] = mnew;
      float p0 = __expf(sc[0][r] - mnew);
      float p1 = __expf(sc[1][r] - mnew);
      sc[0][r] = p0;
      sc[1][r] = p1;
      rs[r] = p0 + p1;
#pragma unroll
      for (int dt = 0; dt < 8; ++dt) acc[dt][r] *= alpha[r];
    }
#pragma unroll
    for (int off = 1; off < 16; off <<= 1)
#pragma unroll
      for (int r = 0; r < 8; ++r) rs[r] += __shfl_xor(rs[r], off, 32);
#pragma unroll
    for (int r = 0; r < 8; ++r) l_i[r] = l_i[r] * alpha[r] + rs[r];

    // P to LDS (C layout) then read back in A-fragment layout
#pragma unroll
    for (int r = 0; r < 8; ++r) {
      int prow = r + 8 * hl;
      Plds[prow * 32 + cn] = sc[0][r];
      Plds[prow * 32 + 16 + cn] = sc[1][r];
    }
    __syncthreads();

    v16bf Pa;
#pragma unroll
    for (int e = 0; e < 16; ++e) Pa[e] = (bf16)Plds[cn * 32 + kmap(e, hl)];

    // P (16x32) @ V (32x128): 8 d-tiles
#pragma unroll
    for (int dt = 0; dt < 8; ++dt) {
      v16bf Vb;
#pragma unroll
      for (int e = 0; e < 16; ++e)
        Vb[e] = Vlds[kmap(e, hl) * 128 + dt * 16 + cn];
      acc[dt] = wmma_bf16(Pa, Vb, acc[dt]);
    }
  }

  float invl[8];
#pragma unroll
  for (int r = 0; r < 8; ++r) invl[r] = 1.0f / l_i[r];
#pragma unroll
  for (int dt = 0; dt < 8; ++dt)
#pragma unroll
    for (int r = 0; r < 8; ++r) {
      size_t o =
          (size_t)((b * S + q0 + r + 8 * hl) * 16 + h) * 128 + dt * 16 + cn;
      ctx[o] = (bf16)(acc[dt][r] * invl[r]);
    }
}

// ---------------------------------------------------------------------------
extern "C" void kernel_launch(void* const* d_in, const int* in_sizes, int n_in,
                              void* d_out, int out_size, void* d_ws,
                              size_t ws_size, hipStream_t stream) {
  (void)in_sizes;
  (void)n_in;
  (void)out_size;
  (void)ws_size;

  const float* x = (const float*)d_in[0];
  const float* Wkv = (const float*)d_in[1];
  const float* Wqd = (const float*)d_in[2];
  const float* Wkc = (const float*)d_in[3];
  const float* Wv = (const float*)d_in[4];
  const float* Wqc = (const float*)d_in[5];
  const float* Wkr = (const float*)d_in[6];
  const float* Wqr = (const float*)d_in[7];
  const float* Wo = (const float*)d_in[8];
  const float* bo = (const float*)d_in[9];
  float* out = (float*)d_out;

  const int B = 2, S = 2048, D = 2048, KVL = 512, QL = 1536;
  const int M = B * S;  // 4096

  char* w = (char*)d_ws;
  bf16* xb = (bf16*)w;     w += (size_t)M * D * 2;
  bf16* Wkv_t = (bf16*)w;  w += (size_t)D * KVL * 2;
  bf16* Wqd_t = (bf16*)w;  w += (size_t)D * QL * 2;
  bf16* Wqc_t = (bf16*)w;  w += (size_t)QL * 1024 * 2;
  bf16* Wqr_t = (bf16*)w;  w += (size_t)QL * 1024 * 2;
  bf16* Wkr_t = (bf16*)w;  w += (size_t)D * 64 * 2;
  bf16* Wkc_t = (bf16*)w;  w += (size_t)KVL * 1024 * 2;
  bf16* Wv_t = (bf16*)w;   w += (size_t)KVL * D * 2;
  bf16* Wo_t = (bf16*)w;   w += (size_t)D * D * 2;
  bf16* c_kv = (bf16*)w;   w += (size_t)M * KVL * 2;
  bf16* c_q = (bf16*)w;    w += (size_t)M * QL * 2;
  bf16* qb = (bf16*)w;     w += (size_t)M * D * 2;
  bf16* kb = (bf16*)w;     w += (size_t)M * D * 2;
  bf16* vb = (bf16*)w;     w += (size_t)M * D * 2;
  bf16* krb = (bf16*)w;    w += (size_t)M * 64 * 2;
  bf16* ctx = (bf16*)w;    w += (size_t)M * D * 2;

  // ---- pack operands (bf16, weights transposed to N x K) ----
  {
    int n = M * D;
    pack_bf16_kernel<<<(n + 255) / 256, 256, 0, stream>>>(x, xb, n);
    auto pw = [&](const float* Wp, bf16* Wt, int K, int N) {
      int e = K * N;
      pack_wt_kernel<<<(e + 255) / 256, 256, 0, stream>>>(Wp, Wt, K, N);
    };
    pw(Wkv, Wkv_t, D, KVL);
    pw(Wqd, Wqd_t, D, QL);
    pw(Wqc, Wqc_t, QL, 1024);
    pw(Wqr, Wqr_t, QL, 1024);
    pw(Wkr, Wkr_t, D, 64);
    pw(Wkc, Wkc_t, KVL, 1024);
    pw(Wv, Wv_t, KVL, D);
    pw(Wo, Wo_t, D, D);
  }

  dim3 blk(128);
  // c_kv = x @ W_kv_down
  gemm_wmma_tdm_kernel<true, false, false>
      <<<dim3(KVL / 64, M / 32), blk, 0, stream>>>(xb, Wkv_t, nullptr, c_kv, M,
                                                   KVL, D, KVL, 0, 0);
  // c_q = x @ W_q_down
  gemm_wmma_tdm_kernel<true, false, false>
      <<<dim3(QL / 64, M / 32), blk, 0, stream>>>(xb, Wqd_t, nullptr, c_q, M,
                                                  QL, D, QL, 0, 0);
  // q_c = c_q @ W_qc -> qb[..., 0:64] per head
  gemm_wmma_tdm_kernel<true, true, false>
      <<<dim3(1024 / 64, M / 32), blk, 0, stream>>>(c_q, Wqc_t, nullptr, qb, M,
                                                    1024, QL, D, 64, 0);
  // q_r (pre-RoPE) = c_q @ W_qr -> qb[..., 64:128] per head
  gemm_wmma_tdm_kernel<true, true, false>
      <<<dim3(1024 / 64, M / 32), blk, 0, stream>>>(c_q, Wqr_t, nullptr, qb, M,
                                                    1024, QL, D, 64, 64);
  // k_r (pre-RoPE) = x @ W_kr -> krb
  gemm_wmma_tdm_kernel<true, false, false>
      <<<dim3(64 / 64, M / 32), blk, 0, stream>>>(xb, Wkr_t, nullptr, krb, M,
                                                  64, D, 64, 0, 0);
  // k_c = c_kv @ W_kc -> kb[..., 0:64] per head
  gemm_wmma_tdm_kernel<true, true, false>
      <<<dim3(1024 / 64, M / 32), blk, 0, stream>>>(c_kv, Wkc_t, nullptr, kb, M,
                                                    1024, KVL, D, 64, 0);
  // v = c_kv @ W_v -> vb
  gemm_wmma_tdm_kernel<true, false, false>
      <<<dim3(D / 64, M / 32), blk, 0, stream>>>(c_kv, Wv_t, nullptr, vb, M, D,
                                                 KVL, D, 0, 0);

  // RoPE in place on q, and RoPE + head-broadcast for k_r
  int tq = M * 16 * 32;
  rope_q_kernel<<<(tq + 255) / 256, 256, 0, stream>>>(qb, tq, S);
  int tk = M * 32;
  rope_k_kernel<<<(tk + 255) / 256, 256, 0, stream>>>(krb, kb, tk, S);

  // attention -> ctx (bf16)
  mla_attn_kernel<<<dim3(S / 16, 16, B), blk.x == 128 ? dim3(32) : dim3(32), 0,
                    stream>>>(qb, kb, vb, ctx, S);

  // out = ctx @ W_o + b_o
  gemm_wmma_tdm_kernel<false, false, true>
      <<<dim3(D / 64, M / 32), blk, 0, stream>>>(ctx, Wo_t, bo, out, M, D, D, D,
                                                 0, 0);
}
